// NeuralMemory_13666585936417
// MI455X (gfx1250) — compile-verified, matrix operand order checked by reading, not run on hardware
//
#include <hip/hip_runtime.h>
#include <hip/hip_bf16.h>

// ---------------------------------------------------------------------------
// NeuralMemory step on MI455X (gfx1250, wave32, WMMA).
//
// Shapes: N = B*S = 32768 tokens, D = 256, L = 2.
// All matmuls run through v_wmma_f32_16x16x32_bf16 (bf16 in, f32 acc).
// GEMM B-tiles (pre-transposed bf16 weights) are moved with
// global_load_async_to_lds_b128 (ASYNCcnt) and double-buffered in LDS so the
// async copies overlap the WMMA stream; A-tiles are f32->bf16 converted while
// staging. Weight grads use split-K WMMA + native f32 global atomics.
// ---------------------------------------------------------------------------

typedef __attribute__((ext_vector_type(16))) __bf16 bf16x16;
typedef __attribute__((ext_vector_type(8)))  float  floatx8;

constexpr int   Dm   = 256;
constexpr int   NTOK = 32768;
constexpr float LRc  = 1e-3f;
constexpr float WDc  = 1e-2f;
constexpr float EPSc = 1e-8f;

// ---- WMMA fragment load from LDS ------------------------------------------
// LDS tile layout: [row][k] with row stride 40 bf16 (80 B, conflict-free,
// 16B-aligned chunks). Per the CDNA5 ISA 16-bit A layout:
//   lane <16 : row = lane,    K = {0..7} then {16..23}
//   lane>=16 : row = lane-16, K = {8..15} then {24..31}
// B tiles are staged TRANSPOSED ([n][k]) so the same loader produces the
// B fragment (columns striped like A rows).
__device__ inline bf16x16 ld_frag(const __bf16* lds, int row) {
  const int lane = threadIdx.x & 31;
  const __bf16* p = lds + (size_t)(row + (lane & 15)) * 40;
  const int hi = (lane >> 4) & 1;
  union { uint4 u[2]; bf16x16 v; } u;
  u.u[0] = *(const uint4*)(p + (hi ? 8  : 0));
  u.u[1] = *(const uint4*)(p + (hi ? 24 : 16));
  return u.v;
}

__device__ inline float sigmoidf_fast(float z) { return 1.f / (1.f + __expf(-z)); }
__device__ inline float silu_grad(float z) {
  float s = sigmoidf_fast(z);
  return s * (1.f + z * (1.f - s));
}

__device__ inline void wait_async_zero() {
  asm volatile("s_wait_asynccnt 0x0" ::: "memory");
}

// ---- Main GEMM: out[N,256] = A[N,256] @ W, epilogue fused -------------------
// WT is the bf16 weight laid out [n][k] (i.e. W transposed) so B staging is a
// straight async DMA copy. FLAGS: 1=+bias[n], 2=SiLU, 4=+addend[idx], 8=store z.
template<int FLAGS>
__global__ __launch_bounds__(256) void gemm_act(
    const float*  __restrict__ A,
    const __bf16* __restrict__ WT,
    const float*  __restrict__ bias,
    const float*  __restrict__ addend,
    float*        __restrict__ zout,
    float*        __restrict__ out) {
  __shared__ __attribute__((aligned(16))) __bf16 lA[2][128][40];
  __shared__ __attribute__((aligned(16))) __bf16 lB[2][128][40];
  const int tid  = threadIdx.x;
  const int lane = tid & 31;
  const int wave = tid >> 5;
  const int wm   = wave & 3;   // 4 strips of 32 rows
  const int wn   = wave >> 2;  // 2 strips of 64 cols
  const int mBase = blockIdx.x * 128;
  const int nBase = blockIdx.y * 128;
  const int srow = tid >> 1;          // staging row (0..127)
  const int skh  = (tid & 1) * 16;    // staging k-half

  floatx8 acc[2][4] = {};
  float fa[16];

  // Async DMA of 32 B of bf16 weights straight into LDS (IOFFSET applies to
  // both LDS and global addresses on async-to-LDS ops).
  auto issueB = [&](int buf, int k0) {
    unsigned loff = (unsigned)(uintptr_t)(&lB[buf][srow][skh]);
    const __bf16* g = WT + (size_t)(nBase + srow) * Dm + k0 + skh;
    asm volatile("global_load_async_to_lds_b128 %0, %1, off\n\t"
                 "global_load_async_to_lds_b128 %0, %1, off offset:16"
                 :: "v"(loff), "v"(g) : "memory");
  };
  auto loadA = [&](int k0) {
    const float* s = A + (size_t)(mBase + srow) * Dm + k0 + skh;
    #pragma unroll
    for (int i = 0; i < 16; ++i) fa[i] = s[i];
  };
  auto storeA = [&](int buf) {
    __bf16 t[16] __attribute__((aligned(16)));
    #pragma unroll
    for (int i = 0; i < 16; ++i) t[i] = (__bf16)fa[i];
    *(uint4*)(&lA[buf][srow][skh])     = *(const uint4*)(&t[0]);
    *(uint4*)(&lA[buf][srow][skh + 8]) = *(const uint4*)(&t[8]);
  };
  auto compute = [&](int buf) {
    bf16x16 af[2], bfr[4];
    #pragma unroll
    for (int i = 0; i < 2; ++i) af[i]  = ld_frag(&lA[buf][0][0], wm * 32 + i * 16);
    #pragma unroll
    for (int j = 0; j < 4; ++j) bfr[j] = ld_frag(&lB[buf][0][0], wn * 64 + j * 16);
    #pragma unroll
    for (int i = 0; i < 2; ++i)
      #pragma unroll
      for (int j = 0; j < 4; ++j)
        acc[i][j] = __builtin_amdgcn_wmma_f32_16x16x32_bf16(
            false, af[i], false, bfr[j], (short)0, acc[i][j], false, false);
  };

  // Prologue: stage tile 0.
  issueB(0, 0);
  loadA(0);
  storeA(0);
  wait_async_zero();
  __syncthreads();

  // Pipelined main loop: stage k+1 while computing k; one barrier per step.
  int cur = 0;
  for (int ks = 1; ks < Dm / 32; ++ks) {
    const int nxt = cur ^ 1;
    issueB(nxt, ks * 32);
    loadA(ks * 32);
    compute(cur);
    storeA(nxt);
    wait_async_zero();
    __syncthreads();
    cur = nxt;
  }
  compute(cur);

  // Epilogue. C layout: VGPR r of lane l -> row = 16tile + r + (l>=16 ? 8:0),
  // col = 16tile + (l&15).
  #pragma unroll
  for (int i = 0; i < 2; ++i) {
    const int rbase = mBase + wm * 32 + i * 16 + ((lane >> 4) ? 8 : 0);
    #pragma unroll
    for (int j = 0; j < 4; ++j) {
      const int col = nBase + wn * 64 + j * 16 + (lane & 15);
      #pragma unroll
      for (int r = 0; r < 8; ++r) {
        const size_t idx = (size_t)(rbase + r) * Dm + col;
        float z = acc[i][j][r];
        if (FLAGS & 1) z += bias[col];
        if (FLAGS & 8) zout[idx] = z;
        float v = z;
        if (FLAGS & 2) v = z * sigmoidf_fast(z);
        if (FLAGS & 4) v += addend[idx];
        out[idx] = v;
      }
    }
  }
}

// ---- Weight-grad GEMM: gW[m][n] += sum_k A[k][m] * B[k][n] ------------------
// Split-K over 16 slices of 2048 tokens; both operands stage-transposed and
// f32->bf16 converted into LDS; f32 atomic reduction into gW.
__global__ __launch_bounds__(256) void wgrad(
    const float* __restrict__ Aact,
    const float* __restrict__ Bact,
    float*       __restrict__ gW) {
  __shared__ __bf16 lA[64][40];
  __shared__ __bf16 lB[64][40];
  const int tid  = threadIdx.x;
  const int lane = tid & 31;
  const int wave = tid >> 5;
  const int wm = wave & 3;   // 4 strips of 16 rows
  const int wn = wave >> 2;  // 2 strips of 32 cols
  const int m0 = blockIdx.x * 64;
  const int n0 = blockIdx.y * 64;
  const int t0 = blockIdx.z * 2048;

  floatx8 acc[2] = {};

  for (int kk = 0; kk < 2048; kk += 32) {
    __syncthreads();
    {
      const int tok = tid >> 3;
      const int fc  = (tid & 7) * 8;
      const float* sa = Aact + (size_t)(t0 + kk + tok) * Dm + m0 + fc;
      const float* sb = Bact + (size_t)(t0 + kk + tok) * Dm + n0 + fc;
      #pragma unroll
      for (int i = 0; i < 8; ++i) lA[fc + i][tok] = (__bf16)sa[i];
      #pragma unroll
      for (int i = 0; i < 8; ++i) lB[fc + i][tok] = (__bf16)sb[i];
    }
    __syncthreads();

    bf16x16 af = ld_frag(&lA[0][0], wm * 16);
    #pragma unroll
    for (int j = 0; j < 2; ++j) {
      bf16x16 bfr = ld_frag(&lB[0][0], wn * 32 + j * 16);
      acc[j] = __builtin_amdgcn_wmma_f32_16x16x32_bf16(
          false, af, false, bfr, (short)0, acc[j], false, false);
    }
  }

  const int rbase = m0 + wm * 16 + ((lane >> 4) ? 8 : 0);
  #pragma unroll
  for (int j = 0; j < 2; ++j) {
    const int col = n0 + wn * 32 + j * 16 + (lane & 15);
    #pragma unroll
    for (int r = 0; r < 8; ++r)
      unsafeAtomicAdd(&gW[(size_t)(rbase + r) * Dm + col], acc[j][r]);
  }
}

// ---- Small kernels ----------------------------------------------------------
__global__ void zero_k(float* p, int n) {
  int i = blockIdx.x * blockDim.x + threadIdx.x;
  if (i < n) p[i] = 0.f;
}

// dst[n*256+k] = (bf16) (transpose ? src[k*256+n] : src[n*256+k])
__global__ void cvt_mat(const float* __restrict__ src, __bf16* __restrict__ dst,
                        int transpose) {
  int i = blockIdx.x * blockDim.x + threadIdx.x;  // 65536
  int n = i >> 8, k = i & 255;
  dst[i] = (__bf16)(transpose ? src[k * 256 + n] : src[i]);
}

// one wave per token: alr = sigmoid(x . Wlr + blr) * 0.1
__global__ void alr_kernel(const float* __restrict__ x, const float* __restrict__ Wlr,
                           const float* __restrict__ blr, float* __restrict__ alr) {
  const int tok  = (blockIdx.x * blockDim.x + threadIdx.x) >> 5;
  const int lane = threadIdx.x & 31;
  const float* xp = x + (size_t)tok * Dm + lane * 8;
  const float* wp = Wlr + lane * 8;
  float s = 0.f;
  #pragma unroll
  for (int i = 0; i < 8; ++i) s += xp[i] * wp[i];
  #pragma unroll
  for (int off = 16; off > 0; off >>= 1) s += __shfl_xor(s, off, 32);
  if (lane == 0) alr[tok] = 0.1f * sigmoidf_fast(s + blr[0]);
}

// dpreds = (2/D)*alr*(preds - values); dz2 = dpreds * silu'(z2)
__global__ void ew_dback2(const float* __restrict__ preds, const float* __restrict__ values,
                          const float* __restrict__ alr, const float* __restrict__ z2,
                          float* __restrict__ dpreds, float* __restrict__ dz2, int total) {
  int i = blockIdx.x * blockDim.x + threadIdx.x;
  if (i >= total) return;
  int t = i >> 8;
  float dp = (2.0f / 256.0f) * alr[t] * (preds[i] - values[i]);
  dpreds[i] = dp;
  dz2[i] = dp * silu_grad(z2[i]);
}

// dz1 = dh1 * silu'(z1)  (in place on dh1 buffer)
__global__ void ew_dz1(float* __restrict__ dh1, const float* __restrict__ z1, int total) {
  int i = blockIdx.x * blockDim.x + threadIdx.x;
  if (i >= total) return;
  dh1[i] *= silu_grad(z1[i]);
}

// gb[n] += sum over 256-token slice of src[t][n]
__global__ void colsum(const float* __restrict__ src, float* __restrict__ dst) {
  const int n = threadIdx.x;
  const size_t t0 = (size_t)blockIdx.x * 256;
  float s = 0.f;
  for (int t = 0; t < 256; ++t) s += src[(t0 + t) * Dm + n];
  unsafeAtomicAdd(&dst[n], s);
}

// AdamW step-1 from zero state + emit -g to d_out
__global__ void adamw_k(const float* __restrict__ p, const float* __restrict__ g,
                        float* __restrict__ np, float* __restrict__ outneg, int n) {
  int i = blockIdx.x * blockDim.x + threadIdx.x;
  if (i >= n) return;
  float gi = g[i];
  np[i] = p[i] * (1.0f - LRc * WDc) - LRc * gi / (fabsf(gi) + EPSc);
  outneg[i] = -gi;
}

// ---------------------------------------------------------------------------
extern "C" void kernel_launch(void* const* d_in, const int* in_sizes, int n_in,
                              void* d_out, int out_size, void* d_ws, size_t ws_size,
                              hipStream_t stream) {
  const float* x   = (const float*)d_in[0];
  const float* Wk  = (const float*)d_in[1];
  const float* Wq  = (const float*)d_in[2];
  const float* Wv  = (const float*)d_in[3];
  const float* Wlr = (const float*)d_in[4];
  const float* blr = (const float*)d_in[5];
  const float* Ws  = (const float*)d_in[6];   // [2,256,256]
  const float* bs  = (const float*)d_in[7];   // [2,256]
  float* out = (float*)d_out;                 // retrieved | -gWs | -gbs

  const size_t NDsz = (size_t)NTOK * Dm;      // 8,388,608
  float* ws      = (float*)d_ws;
  float* keys    = ws;
  float* queries = ws + 1 * NDsz;
  float* values  = ws + 2 * NDsz;
  float* z1      = ws + 3 * NDsz;
  float* h1      = ws + 4 * NDsz;
  float* z2      = ws + 5 * NDsz;
  float* preds   = ws + 6 * NDsz;   // becomes dpreds in place
  float* dz2     = ws + 7 * NDsz;
  float* dh1     = ws + 8 * NDsz;   // becomes dz1 in place
  float* alr     = ws + 9 * NDsz;
  float* gW      = alr + NTOK;              // 2*65536
  float* gb      = gW + 2 * 65536;          // 512
  float* nWs     = gb + 512;                // 131072
  float* nbs     = nWs + 131072;            // 512
  __bf16* bw  = (__bf16*)(nbs + 512);
  __bf16* WkT = bw;
  __bf16* WqT = bw + 1 * 65536;
  __bf16* WvT = bw + 2 * 65536;
  __bf16* W0T = bw + 3 * 65536;
  __bf16* W1T = bw + 4 * 65536;
  __bf16* W1p = bw + 5 * 65536;   // W1 non-transposed (acts as (W1^T)^T for dgrad)
  __bf16* nW0T = bw + 6 * 65536;
  __bf16* nW1T = bw + 7 * 65536;

  const int ND = (int)NDsz;
  const dim3 gg(NTOK / 128, Dm / 128);      // (256, 2)
  const dim3 wg(4, 4, 16);                  // 64x64 tiles x 16 K-splits

  // 0) zero grad accumulators (ws is poisoned by harness)
  zero_k<<<(2 * 65536 + 512 + 255) / 256, 256, 0, stream>>>(gW, 2 * 65536 + 512);

  // 1) weight prep (bf16, [n][k] layouts)
  cvt_mat<<<256, 256, 0, stream>>>(Wk, WkT, 1);
  cvt_mat<<<256, 256, 0, stream>>>(Wq, WqT, 1);
  cvt_mat<<<256, 256, 0, stream>>>(Wv, WvT, 1);
  cvt_mat<<<256, 256, 0, stream>>>(Ws,          W0T, 1);
  cvt_mat<<<256, 256, 0, stream>>>(Ws + 65536,  W1T, 1);
  cvt_mat<<<256, 256, 0, stream>>>(Ws + 65536,  W1p, 0);

  // 2) adaptive lr per token
  alr_kernel<<<NTOK * 32 / 256, 256, 0, stream>>>(x, Wlr, blr, alr);

  // 3) projections
  gemm_act<0><<<gg, 256, 0, stream>>>(x, WkT, nullptr, nullptr, nullptr, keys);
  gemm_act<0><<<gg, 256, 0, stream>>>(x, WqT, nullptr, nullptr, nullptr, queries);
  gemm_act<0><<<gg, 256, 0, stream>>>(x, WvT, nullptr, nullptr, nullptr, values);

  // 4) forward MLP on keys: h = x + silu(x@W + b), keep z for backward
  gemm_act<15><<<gg, 256, 0, stream>>>(keys, W0T, bs,       keys, z1, h1);
  gemm_act<15><<<gg, 256, 0, stream>>>(h1,   W1T, bs + 256, h1,   z2, preds);

  // 5) backward
  ew_dback2<<<(ND + 255) / 256, 256, 0, stream>>>(preds, values, alr, z2, preds, dz2, ND);
  // dh1 = dz2 @ W1^T + dpreds
  gemm_act<4><<<gg, 256, 0, stream>>>(dz2, W1p, nullptr, preds, nullptr, dh1);
  // dz1 = dh1 * silu'(z1)
  ew_dz1<<<(ND + 255) / 256, 256, 0, stream>>>(dh1, z1, ND);
  // gW1 = h1^T @ dz2, gW0 = keys^T @ dz1
  wgrad<<<wg, 256, 0, stream>>>(h1,   dz2, gW + 65536);
  wgrad<<<wg, 256, 0, stream>>>(keys, dh1, gW);
  // gb = column sums
  colsum<<<128, 256, 0, stream>>>(dz2, gb + 256);
  colsum<<<128, 256, 0, stream>>>(dh1, gb);

  // 6) AdamW (step 1, zero state) + surprises (-g) into d_out
  adamw_k<<<(131072 + 255) / 256, 256, 0, stream>>>(Ws, gW, nWs, out + NDsz, 131072);
  adamw_k<<<2, 256, 0, stream>>>(bs, gb, nbs, out + NDsz + 131072, 512);
  cvt_mat<<<256, 256, 0, stream>>>(nWs,         nW0T, 1);
  cvt_mat<<<256, 256, 0, stream>>>(nWs + 65536, nW1T, 1);

  // 7) forward MLP on queries with updated weights -> retrieved (d_out)
  gemm_act<7><<<gg, 256, 0, stream>>>(queries, nW0T, nbs,       queries, nullptr, h1);
  gemm_act<7><<<gg, 256, 0, stream>>>(h1,      nW1T, nbs + 256, h1,      nullptr, out);
}
